// LSTM1_47193100648506
// MI455X (gfx1250) — matrix operand order repaired and checked by reading, not compiled, not run on hardware
//
#include <hip/hip_runtime.h>
#include <stdint.h>

typedef __bf16 bf16;
typedef __attribute__((ext_vector_type(16))) __bf16 v16bf;
typedef __attribute__((ext_vector_type(8)))  __bf16 v8bf;
typedef __attribute__((ext_vector_type(2)))  __bf16 bf16x2;
typedef __attribute__((ext_vector_type(8)))  float  v8f;
typedef __attribute__((ext_vector_type(4)))  unsigned int u32x4;
typedef __attribute__((ext_vector_type(8)))  int i32x8;
typedef __attribute__((ext_vector_type(4)))  int i32x4;

namespace {
constexpr int kBatch  = 64;
constexpr int kSeq    = 512;
constexpr int kEmbed  = 300;
constexpr int kEpad   = 320;   // K padded to multiple of 32 for bf16 WMMA
constexpr int kHidden = 512;
constexpr int kG4     = 2048;  // 4*H
constexpr int kNWG    = 16;    // persistent workgroups in the scan

// workspace layout (bytes)
constexpr size_t OFF_BAR = 0;                                    // barrier counter
constexpr size_t OFF_B   = 1024;                                 // b_ih+b_hh  f32[2048]
constexpr size_t OFF_WIH = 16384;                                // W_ih bf16 padded [2048][320]
constexpr size_t OFF_WHH = OFF_WIH + (size_t)kG4 * kEpad * 2;    // W_hh bf16 [2048][512]
constexpr size_t OFF_H0  = OFF_WHH + (size_t)kG4 * kHidden * 2;  // h ping  bf16 [64][512]
constexpr size_t OFF_H1  = OFF_H0 + (size_t)kBatch * kHidden * 2;// h pong  bf16 [64][512]
constexpr size_t OFF_HT  = OFF_H1 + (size_t)kBatch * kHidden * 2;// final h f32 [64][512]
constexpr size_t OFF_XP  = OFF_HT + (size_t)kBatch * kHidden * 4;// x_proj bf16, tile-swizzled
} // namespace

__device__ __forceinline__ float sigmoid_(float x) { return 1.f / (1.f + __expf(-x)); }

// x_proj tile-swizzled index: tile (s, ctile 0..127, rtile 0..3) stored as
// 32 lanes x 8 bf16 in WMMA C-register order; lane chunk is 16B contiguous.
__device__ __forceinline__ size_t xp_base(int s, int ctile, int rtile, int lane) {
    return (((size_t)s * 128 + ctile) * 4 + rtile) * 256 + lane * 8;
}

// ---------------------------------------------------------------------------
// Init: zero barrier + h buffers, fold biases, convert weights to bf16
// ---------------------------------------------------------------------------
__global__ __launch_bounds__(256) void init_kernel(
    const float* __restrict__ Wih, const float* __restrict__ Whh,
    const float* __restrict__ bih, const float* __restrict__ bhh,
    unsigned* __restrict__ bar, float* __restrict__ bcomb,
    bf16* __restrict__ WihP, bf16* __restrict__ WhhB,
    bf16* __restrict__ h0, bf16* __restrict__ h1)
{
    const int i = blockIdx.x * 256 + threadIdx.x;           // grid covers 2048*512
    if (i < 64) bar[i] = 0u;
    if (i < kG4) bcomb[i] = bih[i] + bhh[i];
    if (i < kG4 * kEpad) {
        const int n = i / kEpad, k = i % kEpad;
        WihP[i] = (k < kEmbed) ? (bf16)Wih[n * kEmbed + k] : (bf16)0.f;
    }
    if (i < kG4 * kHidden) WhhB[i] = (bf16)Whh[i];
    if (i < kBatch * kHidden) { h0[i] = (bf16)0.f; h1[i] = (bf16)0.f; }
}

// ---------------------------------------------------------------------------
// Kernel A: x_proj = embed[x_indx] @ W_ih^T + (b_ih+b_hh), stored tile-swizzled
// One wave -> 16x64 output (A tile reused across 4 WMMA B tiles). bf16 WMMA.
// ---------------------------------------------------------------------------
__global__ __launch_bounds__(256) void embed_proj_gemm(
    const int* __restrict__ xind, const float* __restrict__ emb,
    const bf16* __restrict__ WihP, const float* __restrict__ bcomb,
    bf16* __restrict__ xproj)
{
    const int lane   = threadIdx.x & 31;
    const int waveId = blockIdx.x * 8 + (threadIdx.x >> 5); // 65536 waves
    const int mt     = waveId >> 5;                         // 0..2047 (16-row tiles of M=32768)
    const int nt4    = waveId & 31;                         // 0..31   (64-col groups of N=2048)
    const int nbase  = nt4 * 64;
    const int n15    = lane & 15;
    const int hs     = lane >> 4;                           // lane half select
    const int kbaseA = hs * 8;

    // rows m = mt*16 .. +15 all share timestep s (16 | 64)
    const int s    = mt >> 2;
    const int r    = mt & 3;                                // batch row tile within step
    const int rowA = mt * 16 + n15;                         // this lane's A-matrix row
    const int b    = rowA & 63;
    const int idx  = xind[b * kSeq + s];
    const float* arow = emb + (size_t)idx * kEmbed;

    v8f zero = {};
    v8f acc[4];
#pragma unroll
    for (int t = 0; t < 4; ++t) acc[t] = zero;

    for (int k0 = 0; k0 < kEpad; k0 += 32) {
        // A tile 16x32: lane holds row m, 16 contiguous K (halves split by lane group)
        v16bf a;
#pragma unroll
        for (int rr = 0; rr < 8; ++rr) {
            const int k = k0 + hs * 16 + rr * 2;
            float f0 = 0.f, f1 = 0.f;
            if (k < kEmbed) { float2 f2 = *(const float2*)(arow + k); f0 = f2.x; f1 = f2.y; }
            a[2 * rr]     = (bf16)f0;
            a[2 * rr + 1] = (bf16)f1;
        }
#pragma unroll
        for (int t = 0; t < 4; ++t) {
            const int n = nbase + t * 16 + n15;
            // B tile 32x16: lane holds col n, 16 contiguous K (halves split by lane group)
            const v16bf bt = *(const v16bf*)(WihP + (size_t)n * kEpad + k0 + hs * 16);
            acc[t] = __builtin_amdgcn_wmma_f32_16x16x32_bf16(
                false, a, false, bt, (short)0, acc[t], false, false);
        }
    }

#pragma unroll
    for (int t = 0; t < 4; ++t) {
        const int col  = nbase + t * 16 + n15;
        const float bi = bcomb[col];
        v8bf pk;
#pragma unroll
        for (int v = 0; v < 8; ++v) pk[v] = (bf16)(acc[t][v] + bi);
        *(v8bf*)(xproj + xp_base(s, nt4 * 4 + t, r, lane)) = pk;   // one b128 store
    }
}

// ---------------------------------------------------------------------------
// Kernel B: persistent 16-WG scan. Each WG owns 32 hidden columns; its 128
// W_hh gate rows (128 KB bf16) are brought into LDS by the Tensor Data Mover
// (4 TDM 2D-tile loads, one per gate block) and stay resident for the scan.
// h ping-pongs through L2 with one grid barrier per step. c stays in regs.
// Wave (rt, ct): rows rt*16..+15 (batch), h-cols wg*32+ct*16..+15.
// ---------------------------------------------------------------------------
__global__ __launch_bounds__(256) void lstm_scan(
    const bf16* __restrict__ xproj, const bf16* __restrict__ WhhB,
    bf16* __restrict__ h0, bf16* __restrict__ h1,
    float* __restrict__ hT, unsigned* __restrict__ bar)
{
    extern __shared__ __align__(64) bf16 sWhh[];            // [128][512] bf16 = 128 KB

    const int wg   = blockIdx.x;
    const int lane = threadIdx.x & 31;
    const int wv   = threadIdx.x >> 5;
    const int rt   = wv >> 1;                               // 0..3 batch row tile
    const int ct   = wv & 1;                                // 0..1 col tile within WG
    const int n15  = lane & 15;
    const int hs   = lane >> 4;
    const int kbaseA = hs * 8;

#if __has_builtin(__builtin_amdgcn_tensor_load_to_lds)
    // TDM preload: local rows q*32..q*32+31  <-  W_hh rows q*512 + wg*32 ..+31
    // D# per CDNA5 ISA: 2D tile 512x32, 2-byte elems.
    // This toolchain's builtin arity: (g0 u32x4, g1 i32x8, g2 i32x4, g3 i32x4,
    // extra i32x8, cpol i32); groups 2/3/extra are zero for a 2D tile.
    if (threadIdx.x < 32) {                                 // one wave issues; TENSORcnt per-wave
        const unsigned ldsBase = (unsigned)(uintptr_t)sWhh; // low 32b of flat = LDS offset
        const uint64_t gaBase  = (uint64_t)(uintptr_t)WhhB;
#pragma unroll
        for (int q = 0; q < 4; ++q) {
            const uint64_t ga  = gaBase + ((uint64_t)(q * kHidden + wg * 32) * kHidden) * 2u;
            const unsigned lds = ldsBase + (unsigned)(q * 32 * kHidden * 2);
            u32x4 g0;
            g0.x = 1u;                                           // count=1, user descriptor
            g0.y = lds;                                          // lds_addr (bytes)
            g0.z = (unsigned)(ga & 0xFFFFFFFFu);                 // global_addr[31:0]
            g0.w = (unsigned)((ga >> 32) & 0x1FFFFFFu) | (2u << 30); // addr[56:32] | type=2
            i32x8 g1;
            g1[0] = (int)(1u << 16);                             // wg_mask=0, data_size=2B
            g1[1] = (int)((unsigned)(kHidden & 0xFFFF) << 16);   // tensor_dim0[15:0]=512
            g1[2] = (int)((unsigned)(kG4 & 0xFFFF) << 16);       // dim0 hi=0 | tensor_dim1 lo=2048
            g1[3] = (int)((unsigned)(kHidden & 0xFFFF) << 16);   // dim1 hi=0 | tile_dim0=512
            g1[4] = 32;                                          // tile_dim1=32, tile_dim2=0
            g1[5] = kHidden;                                     // tensor_dim0_stride lo32=512
            g1[6] = 0;                                           // stride hi | dim1_stride lo
            g1[7] = 0;                                           // dim1_stride hi (2D: unused)
            i32x4 gz4 = {0, 0, 0, 0};
            i32x8 gz8 = {0, 0, 0, 0, 0, 0, 0, 0};
            __builtin_amdgcn_tensor_load_to_lds(g0, g1, gz4, gz4, gz8, 0);
        }
        __builtin_amdgcn_s_wait_tensorcnt(0);
    }
    __syncthreads();
#else
    for (int idx = threadIdx.x; idx < 128 * 64; idx += 256) {
        const int lr = idx >> 6, c8 = idx & 63;
        const int q = lr >> 5, j = lr & 31;
        const int grow = q * kHidden + wg * 32 + j;
        *reinterpret_cast<uint4*>(sWhh + (size_t)lr * kHidden + c8 * 8) =
            *reinterpret_cast<const uint4*>(WhhB + (size_t)grow * kHidden + c8 * 8);
    }
    __syncthreads();
#endif

    const int colh   = wg * 32 + ct * 16 + n15;             // h column for this lane
    const int rowA   = rt * 16 + n15;                       // batch row for A operand
    const int ldsrow = ct * 16 + n15;                       // + q*32 selects the gate

    v8f cst = {};                                           // cell state, fp32, in regs

    for (int st = 0; st < kSeq; ++st) {
        const bf16* hread  = (st & 1) ? h1 : h0;
        bf16*       hwrite = (st & 1) ? h0 : h1;

        // gates init = x_proj (bias already folded); tile-swizzled: 1 b128 per gate
        v8f acc[4];
#pragma unroll
        for (int q = 0; q < 4; ++q) {
            const v8bf pk = *(const v8bf*)(xproj + xp_base(st, q * 32 + wg * 2 + ct, rt, lane));
#pragma unroll
            for (int v = 0; v < 8; ++v) acc[q][v] = (float)pk[v];
        }
        if (st + 1 < kSeq)
            __builtin_prefetch(xproj + xp_base(st + 1, wg * 2 + ct, rt, lane), 0, 1);

        // gates += h_{t-1} @ W_hh^T   (A from global h, B from LDS)
        for (int k0 = 0; k0 < kHidden; k0 += 32) {
            v16bf a;
#pragma unroll
            for (int rr = 0; rr < 8; ++rr) {
                const int k = k0 + hs * 16 + rr * 2;
                bf16x2 p = *(const bf16x2*)(hread + (size_t)rowA * kHidden + k);
                a[2 * rr]     = p.x;
                a[2 * rr + 1] = p.y;
            }
#pragma unroll
            for (int q = 0; q < 4; ++q) {
                const v16bf bq = *(const v16bf*)(
                    sWhh + (size_t)(q * 32 + ldsrow) * kHidden + k0 + hs * 16);
                acc[q] = __builtin_amdgcn_wmma_f32_16x16x32_bf16(
                    false, a, false, bq, (short)0, acc[q], false, false);
            }
        }

        // elementwise LSTM cell (PyTorch gate order i,f,g,o), fp32
#pragma unroll
        for (int v = 0; v < 8; ++v) {
            const float ig = sigmoid_(acc[0][v]);
            const float fg = sigmoid_(acc[1][v]);
            const float gg = tanhf(acc[2][v]);
            const float og = sigmoid_(acc[3][v]);
            const float c  = fg * cst[v] + ig * gg;
            cst[v] = c;
            const float h = og * tanhf(c);
            const int mrow = rt * 16 + hs * 8 + v;
            hwrite[(size_t)mrow * kHidden + colh] = (bf16)h;
            if (st == kSeq - 1) hT[(size_t)mrow * kHidden + colh] = h;
        }

        // device-wide barrier across the 16 persistent WGs (monotonic counter)
        __threadfence();
        __syncthreads();
        if (threadIdx.x == 0) {
            __hip_atomic_fetch_add(bar, 1u, __ATOMIC_RELEASE, __HIP_MEMORY_SCOPE_AGENT);
            const unsigned target = (unsigned)(kNWG * (st + 1));
            while (__hip_atomic_load(bar, __ATOMIC_ACQUIRE, __HIP_MEMORY_SCOPE_AGENT) < target)
                __builtin_amdgcn_s_sleep(1);
        }
        __syncthreads();
        __threadfence();
    }
}

// ---------------------------------------------------------------------------
// Kernel C: out[b] = dot(hT[b,:], W_out) + b_out     (tiny)
// ---------------------------------------------------------------------------
__global__ void lstm_out(const float* __restrict__ hT, const float* __restrict__ Wout,
                         const float* __restrict__ bout, float* __restrict__ out)
{
    const int b = threadIdx.x;
    if (b < kBatch) {
        float s = 0.f;
        for (int j = 0; j < kHidden; ++j) s += hT[b * kHidden + j] * Wout[j];
        out[b] = s + bout[0];
    }
}

// ---------------------------------------------------------------------------
extern "C" void kernel_launch(void* const* d_in, const int* in_sizes, int n_in,
                              void* d_out, int out_size, void* d_ws, size_t ws_size,
                              hipStream_t stream)
{
    const int*   xind = (const int*)d_in[0];
    const float* emb  = (const float*)d_in[1];
    const float* Wih  = (const float*)d_in[2];
    const float* Whh  = (const float*)d_in[3];
    const float* bih  = (const float*)d_in[4];
    const float* bhh  = (const float*)d_in[5];
    const float* Wout = (const float*)d_in[6];
    const float* bout = (const float*)d_in[7];
    float* out = (float*)d_out;

    char* ws = (char*)d_ws;
    unsigned* bar = (unsigned*)(ws + OFF_BAR);
    float* bcomb  = (float*)(ws + OFF_B);
    bf16*  WihP   = (bf16*)(ws + OFF_WIH);
    bf16*  WhhB   = (bf16*)(ws + OFF_WHH);
    bf16*  h0     = (bf16*)(ws + OFF_H0);
    bf16*  h1     = (bf16*)(ws + OFF_H1);
    float* hTbuf  = (float*)(ws + OFF_HT);
    bf16*  xproj  = (bf16*)(ws + OFF_XP);

    init_kernel<<<4096, 256, 0, stream>>>(Wih, Whh, bih, bhh, bar, bcomb, WihP, WhhB, h0, h1);
    embed_proj_gemm<<<8192, 256, 0, stream>>>(xind, emb, WihP, bcomb, xproj);
    lstm_scan<<<kNWG, 256, 128 * 1024, stream>>>(xproj, WhhB, h0, h1, hTbuf, bar);
    lstm_out<<<1, 64, 0, stream>>>(hTbuf, Wout, bout, out);
}
